// HeteroSceneGAT_45921790328937
// MI455X (gfx1250) — matrix operand order repaired and checked by reading, not compiled, not run on hardware
//
#include <hip/hip_runtime.h>
#include <hip/hip_bf16.h>
#include <math.h>

// ---------------------------------------------------------------------------
// HeteroSceneGAT fused kernel for gfx1250 (MI455X), wave32.
// One workgroup (256 thr = 8 waves) per batch row.
//   Phase 0: tokens -> LDS via TDM (tensor_load_to_lds, TENSORcnt) with
//            double-buffered staging; fp32 -> (+te) -> bf16 convert overlapped.
//   scores = T @ P        via v_wmma_f32_16x16x32_bf16   (K = d, bf16)
//   U      = attn^T @ T   via v_wmma_f32_16x16x4_f32     (K = n, fp32)
// Everything else (q, P, bias, softmax, message, Wo, LN, FFN, LN) is VALU.
// ---------------------------------------------------------------------------

typedef __attribute__((ext_vector_type(16))) __bf16 v16bf;
typedef __attribute__((ext_vector_type(8)))  __bf16 v8bf;
typedef __attribute__((ext_vector_type(4)))  __bf16 v4bf;
typedef __attribute__((ext_vector_type(8)))  float  v8f;
typedef __attribute__((ext_vector_type(2)))  float  v2f;
typedef unsigned int v4u  __attribute__((ext_vector_type(4)));
typedef int          v8i_ __attribute__((ext_vector_type(8)));
typedef int          v4i_ __attribute__((ext_vector_type(4)));

#define B_    1024
#define D_    256
#define H_    8
#define HD_   32
#define NTOK  257          // 1 ego + 64 obs + 128 hard + 64 soft
#define NP    272          // padded to 17 WMMA M-tiles
#define TP    264          // sT pitch (bf16 elems): 528B/row -> conflict-free tile loads
#define SCP   36           // scores pitch (f32)
#define AP    276          // attn  pitch (f32): 20*m mod 64 pattern -> conflict-free
#define UP    268          // U     pitch (f32)
#define SCALE 0.17677669529663687f   // 1/sqrt(32)
#define NINF  (-__builtin_inff())

// ---- manual LDS partition (staging aliases scores+attn, dead in phase 0) ---
#define OFF_T     0u            // bf16[272*264]          = 143616 B
#define OFF_A     143616u       // f32 [272*36] scores|U  =  39168 B
#define OFF_ATTN  182784u       // f32 [32*276] attn^T    =  35328 B
#define OFF_STG0  143616u       // staging buf0 (32 KB, aliases sA)
#define OFF_STG1  176384u       // staging buf1 (32 KB, aliases sA|sAttn)
#define OFF_PT    218112u       // bf16[32*264] P         =  16896 B
#define OFF_Q     235008u       // f32 [256]
#define OFF_BIAS  236032u       // f32 [32]
#define OFF_MASK  236160u       // f32 [272]
#define OFF_MSG   237248u       // f32 [256]
#define OFF_FUSED 238272u       // f32 [256]
#define OFF_H1    239296u       // f32 [512]
#define OFF_RED   241344u       // f32 [16]
#define SMEM_BYTES 241408u

__device__ __forceinline__ int rowType(int n) {
    return (n == 0) ? 0 : (n < 65) ? 1 : (n < 193) ? 2 : 3;
}

__device__ __forceinline__ float waveRed(float v) {
#pragma unroll
    for (int i = 16; i > 0; i >>= 1) v += __shfl_xor(v, i, 32);
    return v;
}

__device__ __forceinline__ float waveMax(float v) {
#pragma unroll
    for (int i = 16; i > 0; i >>= 1) v = fmaxf(v, __shfl_xor(v, i, 32));
    return v;
}

__device__ __forceinline__ v16bf ld16(const __bf16* p0, const __bf16* p1) {
    v8bf lo = *(const v8bf*)p0;
    v8bf hi = *(const v8bf*)p1;
    v16bf r;
#pragma unroll
    for (int i = 0; i < 8; ++i) { r[i] = lo[i]; r[i + 8] = hi[i]; }
    return r;
}

__device__ __forceinline__ v8f wmma4(v2f a, v2f b, v8f c) {
    return __builtin_amdgcn_wmma_f32_16x16x4_f32(false, a, false, b, (short)0, c,
                                                 false, false);
}

__device__ __forceinline__ float gelu_exact(float x) {
    return 0.5f * x * (1.f + erff(x * 0.70710678118654752f));
}

// Issue one TDM transfer: rows x 256 fp32, contiguous (stride 256), -> LDS.
// D# per CDNA5 ISA §8.3/8.4: group0 = {count=1 | lds_addr | global_addr | type=2},
// group1 = {data_size=4B, tensor_dim0=256, tensor_dim1=rows, tile_dim0=256,
//           tile_dim1=rows, tensor_dim0_stride=256}.  2D -> groups 2/3 zero.
__device__ __forceinline__ void tdm_load_rows(const float* gsrc, unsigned lds_off,
                                              int rows) {
    unsigned long long ga = (unsigned long long)(__UINTPTR_TYPE__)gsrc;
    v4u g0 = { 1u,                                     // count=1, user mode
               lds_off,                                // LDS byte address
               (unsigned)ga,                           // global_addr[31:0]
               (unsigned)((ga >> 32) & 0x1FFFFFFu) | 0x80000000u }; // [56:32]|type=2
    v8i_ g1 = { (int)0x20000u,                         // data_size=2 (4B)
                (int)(256u << 16),                     // tensor_dim0 = 256
                (int)((unsigned)rows << 16),           // tensor_dim1 = rows
                (int)(256u << 16),                     // tile_dim0 = 256
                rows,                                  // tile_dim1 = rows
                256, 0, 0 };                           // tensor_dim0_stride = 256
    v4i_ z4 = { 0, 0, 0, 0 };
#if defined(__clang_major__) && (__clang_major__ >= 23)
    v8i_ z8 = { 0, 0, 0, 0, 0, 0, 0, 0 };
    __builtin_amdgcn_tensor_load_to_lds(g0, g1, z4, z4, z8, 0);
#else
    __builtin_amdgcn_tensor_load_to_lds(g0, g1, z4, z4, 0);
#endif
}

__global__ __launch_bounds__(256, 1)
void hetero_gat_fused(const float* __restrict__ ego,
                      const float* __restrict__ obs,
                      const float* __restrict__ hard,
                      const float* __restrict__ soft,
                      const int*   __restrict__ omask,
                      const int*   __restrict__ hmask,
                      const int*   __restrict__ smask,
                      const float* __restrict__ te,
                      const float* __restrict__ Wq,
                      const float* __restrict__ Wk,
                      const float* __restrict__ Wv,
                      const float* __restrict__ relk,
                      const float* __restrict__ relb,
                      const float* __restrict__ Wo,
                      const float* __restrict__ bo,
                      const float* __restrict__ ln1_g,
                      const float* __restrict__ ln1_b,
                      const float* __restrict__ W1,
                      const float* __restrict__ b1,
                      const float* __restrict__ W2,
                      const float* __restrict__ b2,
                      const float* __restrict__ ln2_g,
                      const float* __restrict__ ln2_b,
                      float*       __restrict__ out)
{
    __shared__ __align__(16) unsigned char smem[SMEM_BYTES];
    __bf16* sT    = (__bf16*)(smem + OFF_T);
    float*  sA    = (float*) (smem + OFF_A);
    float*  sAttn = (float*) (smem + OFF_ATTN);
    __bf16* sPt   = (__bf16*)(smem + OFF_PT);
    float*  sQ    = (float*) (smem + OFF_Q);
    float*  sBias = (float*) (smem + OFF_BIAS);
    float*  sMask = (float*) (smem + OFF_MASK);
    float*  sMsg  = (float*) (smem + OFF_MSG);
    float*  sFused= (float*) (smem + OFF_FUSED);
    float*  sH1   = (float*) (smem + OFF_H1);
    float*  sRed  = (float*) (smem + OFF_RED);
    float*  sU    = sA;                         // U[32][UP] aliases scores

    const int tid  = threadIdx.x;
    const int wave = tid >> 5;
    const int lane = tid & 31;
    const int b    = blockIdx.x;

    // ===== Phase 0: TDM token staging + (+te, bf16) conversion ===============
    // 9 chunks, each within one token type (boundaries at rows 1/65/193).
    const int c_row0[9] = { 0, 1, 33, 65, 97, 129, 161, 193, 225 };
    const int c_rows[9] = { 1, 32, 32, 32, 32, 32, 32, 32, 32 };
    const int c_type[9] = { 0, 1, 1, 2, 2, 2, 2, 3, 3 };
    #define CHUNK_SRC(i_)                                                      \
        ((c_type[i_] == 0) ? ego  + (size_t)b * D_                             \
       : (c_type[i_] == 1) ? obs  + ((size_t)b * 64  + (c_row0[i_] - 1))   * D_\
       : (c_type[i_] == 2) ? hard + ((size_t)b * 128 + (c_row0[i_] - 65))  * D_\
       :                     soft + ((size_t)b * 64  + (c_row0[i_] - 193)) * D_)

    if (tid < 32)   // one wave issues DMA #0 (per-wave op, EXEC ignored)
        tdm_load_rows(CHUNK_SRC(0), OFF_STG0, c_rows[0]);

    // overlap DMA latency with pad-row zeroing and mask construction
    for (int i = tid; i < (NP - NTOK) * TP; i += 256)
        sT[NTOK * TP + i] = (__bf16)0.f;
    for (int n = tid; n < NP; n += 256) {
        float v;
        if (n == 0)          v = 0.f;
        else if (n < 65)     v = omask[b * 64  + (n - 1)]   ? 0.f : NINF;
        else if (n < 193)    v = hmask[b * 128 + (n - 65)]  ? 0.f : NINF;
        else if (n < NTOK)   v = smask[b * 64  + (n - 193)] ? 0.f : NINF;
        else                 v = NINF;
        sMask[n] = v;
    }

    for (int i = 0; i < 9; ++i) {
        if (tid < 32) {
            if (i + 1 < 9) {   // pipeline: issue next DMA, wait only for current
                tdm_load_rows(CHUNK_SRC(i + 1),
                              ((i + 1) & 1) ? OFF_STG1 : OFF_STG0, c_rows[i + 1]);
                __builtin_amdgcn_s_wait_tensorcnt(1);
            } else {
                __builtin_amdgcn_s_wait_tensorcnt(0);
            }
        }
        __syncthreads();       // chunk i resident; all waves convert it
        const float* stg = (const float*)(smem + ((i & 1) ? OFF_STG1 : OFF_STG0));
        int r0 = c_row0[i], R = c_rows[i], ty = c_type[i];
        for (int idx = tid; idx < R * 64; idx += 256) {
            int r = idx >> 6, c4 = idx & 63;
            float4 v  = ((const float4*)stg)[idx];
            float4 tv = ((const float4*)te)[ty * 64 + c4];
            v4bf pk = { (__bf16)(v.x + tv.x), (__bf16)(v.y + tv.y),
                        (__bf16)(v.z + tv.z), (__bf16)(v.w + tv.w) };
            *(v4bf*)(&sT[(r0 + r) * TP + c4 * 4]) = pk;
        }
        __syncthreads();       // buffer consumed before it is re-targeted
    }

    // ===== Phase 1: q[c] = (ego+te0) @ Wq ====================================
    {
        float acc = 0.f;
#pragma unroll 8
        for (int d = 0; d < D_; ++d)
            acc += (float)sT[d] * Wq[(size_t)d * D_ + tid];   // sT row 0 broadcast
        sQ[tid] = acc;
    }
    __syncthreads();

    // ===== Phase 2: P[col=(t,h)][d]; attn bias; zero attn region =============
    for (int i = tid; i < 32 * AP; i += 256) sAttn[i] = 0.f;  // staging now dead
    {
        int col = tid >> 3, dg = tid & 7;
        int t = col >> 3, h = col & 7, hb = h * HD_;
        float qreg[HD_];
#pragma unroll
        for (int j = 0; j < HD_; ++j) qreg[j] = sQ[hb + j];
        const float* wk = Wk + (size_t)t * D_ * D_;
        for (int dd = 0; dd < 32; ++dd) {
            int d = dg * 32 + dd;
            const float* w = wk + (size_t)d * D_ + hb;
            float acc = 0.f;
#pragma unroll
            for (int j = 0; j < HD_; ++j) acc += qreg[j] * w[j];
            sPt[col * TP + d] = (__bf16)acc;
        }
        if (dg == 0) {  // bias4[col] = scale*(q_h . relk[t,h]) + relb[t,h]
            const float* rk = relk + ((size_t)t * H_ + h) * HD_;
            float acc = 0.f;
#pragma unroll
            for (int j = 0; j < HD_; ++j) acc += qreg[j] * rk[j];
            sBias[col] = acc * SCALE + relb[t * H_ + h];
        }
    }
    __syncthreads();

    // ===== Phase 3: scores_all[272][32] = T @ P  (bf16 WMMA 16x16x32) ========
    {
        int nA = lane & 15;
        int akoff = (lane < 16) ? 0 : 8;    // A: lanes>=16 hold K {8..15,24..31}
        int bkoff = (lane < 16) ? 0 : 8;    // B: lanes>=16 hold K {8..15,24..31}
        for (int job = wave; job < 34; job += 8) {
            int mt = job >> 1, nt = job & 1;
            int mrow = mt * 16 + nA;
            int bcol = nt * 16 + nA;
            v8f acc = {};
#pragma unroll
            for (int kt = 0; kt < 8; ++kt) {
                int k0 = kt * 32;
                v16bf a = ld16(&sT[mrow * TP + k0 + akoff],
                               &sT[mrow * TP + k0 + 16 + akoff]);
                v16bf bb = ld16(&sPt[bcol * TP + k0 + bkoff],
                                &sPt[bcol * TP + k0 + bkoff + 16]);
                acc = __builtin_amdgcn_wmma_f32_16x16x32_bf16(
                          false, a, false, bb, (short)0, acc, false, false);
            }
            int rbase = mt * 16 + ((lane < 16) ? 0 : 8);
#pragma unroll
            for (int r = 0; r < 8; ++r)
                sA[(rbase + r) * SCP + bcol] = acc[r];
        }
    }
    __syncthreads();

    // ===== Phase 4: masked softmax over n (wave w = head w) ==================
    {
        int h = wave;
        float svals[9];
        float m = NINF;
#pragma unroll
        for (int it = 0; it < 9; ++it) {
            int n = it * 32 + lane;
            float s = NINF;
            if (n < NP) {
                int col = rowType(n) * 8 + h;
                s = sA[n * SCP + col] * SCALE + sBias[col] + sMask[n];
            }
            svals[it] = s;
            m = fmaxf(m, s);
        }
        m = waveMax(m);                      // ego row guarantees m is finite
        float sum = 0.f;
#pragma unroll
        for (int it = 0; it < 9; ++it) {
            int n = it * 32 + lane;
            float e = expf(svals[it] - m);   // exp(-inf)=0 for masked/pad rows
            if (n < NP) sAttn[(rowType(n) * 8 + h) * AP + n] = e;
            sum += e;
        }
        sum = waveRed(sum);
        float inv = 1.f / sum;
#pragma unroll
        for (int it = 0; it < 9; ++it) {
            int n = it * 32 + lane;
            if (n < NP) sAttn[(rowType(n) * 8 + h) * AP + n] *= inv;
        }
    }
    __syncthreads();   // sA dead; safe to alias as U below

    // ===== Phase 5: U[32][256] = attn^T @ T  (fp32 WMMA 16x16x4) =============
    {
        __builtin_prefetch(Wv + (size_t)tid * 1024, 0, 1);   // warm Wv for phase 6
        int d0 = wave * 32;                  // each wave: 2 N-tiles x 2 M-tiles
        int nA = lane & 15;
        int kb = (lane < 16) ? 0 : 2;
        v8f a00 = {}, a01 = {}, a10 = {}, a11 = {};
#pragma unroll 4
        for (int k0 = 0; k0 < NP; k0 += 4) { // pads are zero in sAttn and sT
            const float* ap0 = &sAttn[nA * AP + k0 + kb];
            const float* ap1 = &sAttn[(16 + nA) * AP + k0 + kb];
            v2f a0 = { ap0[0], ap0[1] };
            v2f a1 = { ap1[0], ap1[1] };
            const __bf16* t0 = &sT[(k0 + kb) * TP + d0 + nA];
            const __bf16* t1 = &sT[(k0 + kb + 1) * TP + d0 + nA];
            v2f b0 = { (float)t0[0],  (float)t1[0]  };
            v2f b1 = { (float)t0[16], (float)t1[16] };
            a00 = wmma4(a0, b0, a00);  a01 = wmma4(a0, b1, a01);
            a10 = wmma4(a1, b0, a10);  a11 = wmma4(a1, b1, a11);
        }
        int rb = (lane < 16) ? 0 : 8;
#pragma unroll
        for (int r = 0; r < 8; ++r) {
            sU[(rb + r) * UP      + d0 + nA]      = a00[r];
            sU[(rb + r) * UP      + d0 + 16 + nA] = a01[r];
            sU[(16 + rb + r) * UP + d0 + nA]      = a10[r];
            sU[(16 + rb + r) * UP + d0 + 16 + nA] = a11[r];
        }
    }
    __syncthreads();

    // ===== Phase 6: message[c] = sum_t U[t*8+h] . Wv[t][:,c]  (h = c>>5) =====
    {
        __builtin_prefetch(W1 + (size_t)tid * 512, 0, 1);    // warm W1 for FFN
        int c = tid, h = tid >> 5;
        float acc = 0.f;
        for (int t = 0; t < 4; ++t) {
            const float* u  = &sU[(t * 8 + h) * UP];          // LDS broadcast
            const float* wv = Wv + (size_t)t * D_ * D_ + c;   // coalesced in c
#pragma unroll 8
            for (int d = 0; d < D_; ++d)
                acc += u[d] * wv[(size_t)d * D_];
        }
        sMsg[c] = acc;
    }
    __syncthreads();

    // ===== Phase 7: fused = LN1(ego + message @ Wo + bo) =====================
    {
        __builtin_prefetch(W2 + (size_t)tid * 512, 0, 1);    // warm W2 for FFN
        int c = tid;
        float acc = 0.f;
#pragma unroll 8
        for (int d = 0; d < D_; ++d)
            acc += sMsg[d] * Wo[(size_t)d * D_ + c];
        float x = ego[(size_t)b * D_ + c] + acc + bo[c];
        float s1 = waveRed(x), s2 = waveRed(x * x);
        if (lane == 0) { sRed[wave] = s1; sRed[8 + wave] = s2; }
        __syncthreads();
        float S = 0.f, SS = 0.f;
#pragma unroll
        for (int i = 0; i < 8; ++i) { S += sRed[i]; SS += sRed[8 + i]; }
        float mean = S * (1.f / 256.f);
        float var  = SS * (1.f / 256.f) - mean * mean;
        float rstd = rsqrtf(var + 1e-5f);
        sFused[c] = (x - mean) * rstd * ln1_g[c] + ln1_b[c];
    }
    __syncthreads();

    // ===== Phase 8: FFN + LN2 -> out =========================================
    {
#pragma unroll
        for (int jj = 0; jj < 2; ++jj) {     // each thread: 2 hidden units
            int j = tid + jj * 256;
            float acc = 0.f;
#pragma unroll 8
            for (int d = 0; d < D_; ++d)
                acc += sFused[d] * W1[(size_t)d * (2 * D_) + j];
            sH1[j] = gelu_exact(acc + b1[j]);
        }
    }
    __syncthreads();
    {
        int c = tid;
        float acc = 0.f;
#pragma unroll 8
        for (int j = 0; j < 2 * D_; ++j)
            acc += sH1[j] * W2[(size_t)j * D_ + c];
        float x = sFused[c] + acc + b2[c];
        float s1 = waveRed(x), s2 = waveRed(x * x);
        if (lane == 0) { sRed[wave] = s1; sRed[8 + wave] = s2; }
        __syncthreads();
        float S = 0.f, SS = 0.f;
#pragma unroll
        for (int i = 0; i < 8; ++i) { S += sRed[i]; SS += sRed[8 + i]; }
        float mean = S * (1.f / 256.f);
        float var  = SS * (1.f / 256.f) - mean * mean;
        float rstd = rsqrtf(var + 1e-5f);
        out[(size_t)b * D_ + c] = (x - mean) * rstd * ln2_g[c] + ln2_b[c];
    }
}

extern "C" void kernel_launch(void* const* d_in, const int* in_sizes, int n_in,
                              void* d_out, int out_size, void* d_ws, size_t ws_size,
                              hipStream_t stream) {
    (void)in_sizes; (void)n_in; (void)out_size; (void)d_ws; (void)ws_size;
    const float* ego   = (const float*)d_in[0];
    const float* obs   = (const float*)d_in[1];
    const float* hard  = (const float*)d_in[2];
    const float* soft  = (const float*)d_in[3];
    const int*   omask = (const int*)  d_in[4];
    const int*   hmask = (const int*)  d_in[5];
    const int*   smask = (const int*)  d_in[6];
    const float* te    = (const float*)d_in[7];
    const float* Wq    = (const float*)d_in[8];
    const float* Wk    = (const float*)d_in[9];
    const float* Wv    = (const float*)d_in[10];
    const float* relk  = (const float*)d_in[11];
    const float* relb  = (const float*)d_in[12];
    const float* Wo    = (const float*)d_in[13];
    const float* bo    = (const float*)d_in[14];
    const float* ln1g  = (const float*)d_in[15];
    const float* ln1b  = (const float*)d_in[16];
    const float* W1    = (const float*)d_in[17];
    const float* b1    = (const float*)d_in[18];
    const float* W2    = (const float*)d_in[19];
    const float* b2    = (const float*)d_in[20];
    const float* ln2g  = (const float*)d_in[21];
    const float* ln2b  = (const float*)d_in[22];
    float* outp = (float*)d_out;

    hetero_gat_fused<<<dim3(B_), dim3(256), 0, stream>>>(
        ego, obs, hard, soft, omask, hmask, smask, te, Wq, Wk, Wv, relk, relb,
        Wo, bo, ln1g, ln1b, W1, b1, W2, b2, ln2g, ln2b, outp);
}
